// MultiheadAttention_69140383531279
// MI455X (gfx1250) — matrix operand order repaired and checked by reading, not compiled, not run on hardware
//
#include <hip/hip_runtime.h>

// ---------------------------------------------------------------------------
// CDNA5 (gfx1250) fused multihead attention, wave32 + v_wmma_f32_16x16x32_bf16
// - one-time fp32->bf16 conversion pass (weights + activations)
// - all-bf16 WMMA GEMMs with fp32 accumulation
// - 32x64 output tile per wave in GEMMs (2x A-reuse per B fragment set)
// - single-step guarded double buffering: empirically the best codegen
//   (clause-batched b128 loads, staggered loadcnt waits, <256 VGPRs)
// ---------------------------------------------------------------------------
typedef __attribute__((ext_vector_type(16))) __bf16 v16bf;
typedef __attribute__((ext_vector_type(8)))  float  v8f;

#define TDIM 4096
#define SDIM 4096
#define EDIM 1024
#define HEADS 16
#define HD 64

__device__ __forceinline__ v8f wmma_bf16(v16bf a, v16bf b, v8f c) {
  return __builtin_amdgcn_wmma_f32_16x16x32_bf16(
      /*neg_a=*/false, a, /*neg_b=*/false, b,
      /*c_mod=*/(short)0, c, /*reuse_a=*/false, /*reuse_b=*/false);
}

// A fragment (16x32 bf16, M x K), row-major source with leading dim `ld`.
// ISA layout: lane L holds M=L&15, K in {g..g+7, 16+g..16+g+7}, g=(L>>4)*8.
__device__ __forceinline__ v16bf load_A_bf16(const __bf16* base, int ld, int lane) {
  int m = lane & 15;
  int g = (lane >> 4) << 3;
  const __bf16* r = base + (size_t)m * ld;
  union { v16bf v; uint4 q[2]; } f;
  f.q[0] = *reinterpret_cast<const uint4*>(r + g);
  f.q[1] = *reinterpret_cast<const uint4*>(r + 16 + g);
  return f.v;
}

// B fragment (32x16 bf16, K x N) where B[k][n] = src[nBase+n][kBase+k]
// (transposed view; src rows contiguous in k). Lane holds column n=lane&15,
// K in {kq..kq+7, 16+kq..16+kq+7}, kq=(lane>>4)*8.
__device__ __forceinline__ v16bf load_Bt_bf16(const __bf16* src, int ld,
                                              int nBase, int kBase, int lane) {
  int n  = lane & 15;
  int kq = (lane >> 4) << 3;
  const __bf16* r = src + (size_t)(nBase + n) * ld + kBase + kq;
  union { v16bf v; uint4 q[2]; } f;
  f.q[0] = *reinterpret_cast<const uint4*>(r);
  f.q[1] = *reinterpret_cast<const uint4*>(r + 16);
  return f.v;
}

// ---------------------------------------------------------------------------
// Kernel 0: one-time fp32 -> bf16 conversion (8 elems/thread, b128 in/out).
// ---------------------------------------------------------------------------
__global__ void __launch_bounds__(256)
cvt_bf16_kernel(const float* __restrict__ src, __bf16* __restrict__ dst, int n) {
  int i = (blockIdx.x * 256 + threadIdx.x) * 8;
  if (i >= n) return;
  const float4* s = reinterpret_cast<const float4*>(src + i);
  float4 x0 = s[0];
  float4 x1 = s[1];
  union { __bf16 b[8]; uint4 q; } u;
  u.b[0] = (__bf16)x0.x; u.b[1] = (__bf16)x0.y;
  u.b[2] = (__bf16)x0.z; u.b[3] = (__bf16)x0.w;
  u.b[4] = (__bf16)x1.x; u.b[5] = (__bf16)x1.y;
  u.b[6] = (__bf16)x1.z; u.b[7] = (__bf16)x1.w;
  *reinterpret_cast<uint4*>(dst + i) = u.q;
}

// ---------------------------------------------------------------------------
// GEMM core: per-wave 32(M) x 64(N) tile over K=EDIM. One K-step lookahead
// with guarded rotation — the form the backend schedules best (round-3
// codegen: s_clause load batches, staggered waits, no VGPR-MSB spill-over).
// ---------------------------------------------------------------------------
__device__ __forceinline__ void gemm_32x64(const __bf16* __restrict__ Xp,
                                           const __bf16* __restrict__ W,
                                           int nBase, int lane, v8f c[2][4]) {
  v16bf a0 = load_A_bf16(Xp, EDIM, lane);
  v16bf a1 = load_A_bf16(Xp + 16 * EDIM, EDIM, lane);
  v16bf b0 = load_Bt_bf16(W, EDIM, nBase +  0, 0, lane);
  v16bf b1 = load_Bt_bf16(W, EDIM, nBase + 16, 0, lane);
  v16bf b2 = load_Bt_bf16(W, EDIM, nBase + 32, 0, lane);
  v16bf b3 = load_Bt_bf16(W, EDIM, nBase + 48, 0, lane);
  for (int k = 32; k <= EDIM; k += 32) {
    v16bf a0n = a0, a1n = a1, b0n = b0, b1n = b1, b2n = b2, b3n = b3;
    if (k < EDIM) {  // issue next-iteration loads before current WMMAs
      a0n = load_A_bf16(Xp + k, EDIM, lane);
      a1n = load_A_bf16(Xp + 16 * EDIM + k, EDIM, lane);
      b0n = load_Bt_bf16(W, EDIM, nBase +  0, k, lane);
      b1n = load_Bt_bf16(W, EDIM, nBase + 16, k, lane);
      b2n = load_Bt_bf16(W, EDIM, nBase + 32, k, lane);
      b3n = load_Bt_bf16(W, EDIM, nBase + 48, k, lane);
    }
    c[0][0] = wmma_bf16(a0, b0, c[0][0]);
    c[0][1] = wmma_bf16(a0, b1, c[0][1]);
    c[0][2] = wmma_bf16(a0, b2, c[0][2]);
    c[0][3] = wmma_bf16(a0, b3, c[0][3]);
    c[1][0] = wmma_bf16(a1, b0, c[1][0]);
    c[1][1] = wmma_bf16(a1, b1, c[1][1]);
    c[1][2] = wmma_bf16(a1, b2, c[1][2]);
    c[1][3] = wmma_bf16(a1, b3, c[1][3]);
    a0 = a0n; a1 = a1n; b0 = b0n; b1 = b1n; b2 = b2n; b3 = b3n;
  }
}

// ---------------------------------------------------------------------------
// Kernel 1: in-projection. Y = X * W^T + b for Q,K,V (3E columns).
// Block: 8 waves, tile 256(M) x 64(N); grid (T/256, 3E/64).
// Q scaled by 1/sqrt(hd), bf16 (T x E); K bf16 (T x E); V transposed (E x T).
// ---------------------------------------------------------------------------
__global__ void __launch_bounds__(256)
inproj_kernel(const __bf16* __restrict__ Xq, const __bf16* __restrict__ Xk,
              const __bf16* __restrict__ Xv, const __bf16* __restrict__ W,
              const float* __restrict__ bias,
              __bf16* __restrict__ Qb, __bf16* __restrict__ Kb,
              __bf16* __restrict__ Vt) {
  int lane = threadIdx.x & 31;
  int wave = threadIdx.x >> 5;
  int mBase = blockIdx.x * 256 + wave * 32;
  int nGlob = blockIdx.y * 64;            // column in [0, 3E)
  int third = nGlob / EDIM;               // 0=q 1=k 2=v
  int n0    = nGlob % EDIM;
  const __bf16* X = (third == 0) ? Xq : (third == 1) ? Xk : Xv;
  const __bf16* Xp = X + (size_t)mBase * EDIM;

  v8f c[2][4];
#pragma unroll
  for (int r = 0; r < 2; ++r)
#pragma unroll
    for (int j = 0; j < 4; ++j) c[r][j] = v8f{0};

  gemm_32x64(Xp, W, nGlob, lane, c);

  int col = lane & 15;
#pragma unroll
  for (int r = 0; r < 2; ++r) {
    int row = mBase + r * 16 + ((lane >> 4) << 3);  // C layout: vgpr i -> row+i
#pragma unroll
    for (int j = 0; j < 4; ++j) {
      float bv = bias[nGlob + j * 16 + col];
      if (third == 2) {
        // transposed V store: 8 contiguous bf16 per lane -> one b128 store
        union { __bf16 b[8]; uint4 q; } u;
#pragma unroll
        for (int i = 0; i < 8; ++i) u.b[i] = (__bf16)(c[r][j][i] + bv);
        *reinterpret_cast<uint4*>(
            &Vt[(size_t)(n0 + j * 16 + col) * TDIM + row]) = u.q;
      } else {
#pragma unroll
        for (int i = 0; i < 8; ++i) {
          float y = c[r][j][i] + bv;
          if (third == 0)
            Qb[(size_t)(row + i) * EDIM + n0 + j * 16 + col] =
                (__bf16)(y * 0.125f);
          else
            Kb[(size_t)(row + i) * EDIM + n0 + j * 16 + col] = (__bf16)y;
        }
      }
    }
  }
}

// ---------------------------------------------------------------------------
// Kernel 2: flash attention per (head, 128-query tile). Online softmax over
// S in 32-column chunks. V fragments issued before the score WMMAs; next
// chunk's K fragments issued before the softmax VALU block. Writes ctx (bf16)
// and per-row (m,l) stats for the attn-average kernel.
// ---------------------------------------------------------------------------
__global__ void __launch_bounds__(256)
flash_attn_kernel(const __bf16* __restrict__ Qb, const __bf16* __restrict__ Kb,
                  const __bf16* __restrict__ Vt, __bf16* __restrict__ Ctx,
                  float* __restrict__ stat_m, float* __restrict__ stat_l) {
  int lane = threadIdx.x & 31;
  int wave = threadIdx.x >> 5;
  int h = blockIdx.y;
  int tBase = blockIdx.x * 128 + wave * 16;

  __shared__ __bf16 pstage[8][16 * 32];   // per-wave P staging (C -> A layout)
  __bf16* ps = pstage[wave];

  const __bf16* Qh = Qb + (size_t)tBase * EDIM + h * HD;
  const __bf16* Kh = Kb + h * HD;                 // row stride EDIM
  const __bf16* Vh = Vt + (size_t)h * HD * TDIM;  // row stride TDIM

  v16bf aq0 = load_A_bf16(Qh, EDIM, lane);        // K = 0..31 of head dim
  v16bf aq1 = load_A_bf16(Qh + 32, EDIM, lane);   // K = 32..63

  float mrow[8], lrow[8];
  v8f o[4] = {v8f{0}, v8f{0}, v8f{0}, v8f{0}};
#pragma unroll
  for (int i = 0; i < 8; ++i) { mrow[i] = -1e30f; lrow[i] = 0.f; }

  // prologue: K fragments for chunk 0
  v16bf bk0 = load_Bt_bf16(Kh, EDIM,  0, 0,  lane);
  v16bf bk1 = load_Bt_bf16(Kh, EDIM,  0, 32, lane);
  v16bf bk2 = load_Bt_bf16(Kh, EDIM, 16, 0,  lane);
  v16bf bk3 = load_Bt_bf16(Kh, EDIM, 16, 32, lane);

  for (int sb = 0; sb < SDIM; sb += 32) {
    // V fragments for the current chunk: in flight during score WMMAs+softmax
    v16bf bv0 = load_Bt_bf16(Vh, TDIM,  0, sb, lane);
    v16bf bv1 = load_Bt_bf16(Vh, TDIM, 16, sb, lane);
    v16bf bv2 = load_Bt_bf16(Vh, TDIM, 32, sb, lane);
    v16bf bv3 = load_Bt_bf16(Vh, TDIM, 48, sb, lane);

    // scores: 16 x 32 tile = two 16x16 WMMA outputs, K over hd=64
    v8f s0 = v8f{0}, s1 = v8f{0};
    s0 = wmma_bf16(aq0, bk0, s0);
    s0 = wmma_bf16(aq1, bk1, s0);
    s1 = wmma_bf16(aq0, bk2, s1);
    s1 = wmma_bf16(aq1, bk3, s1);

    // next chunk's K fragments: in flight during the softmax VALU block
    int sn = (sb + 32 < SDIM) ? sb + 32 : sb;
    bk0 = load_Bt_bf16(Kh, EDIM, sn,      0,  lane);
    bk1 = load_Bt_bf16(Kh, EDIM, sn,      32, lane);
    bk2 = load_Bt_bf16(Kh, EDIM, sn + 16, 0,  lane);
    bk3 = load_Bt_bf16(Kh, EDIM, sn + 16, 32, lane);

    float alpha[8];
#pragma unroll
    for (int i = 0; i < 8; ++i) {
      float mx = fmaxf(s0[i], s1[i]);
#pragma unroll
      for (int d = 1; d < 16; d <<= 1) mx = fmaxf(mx, __shfl_xor(mx, d, 32));
      float mn = fmaxf(mrow[i], mx);
      alpha[i] = __expf(mrow[i] - mn);
      float p0 = __expf(s0[i] - mn);
      float p1 = __expf(s1[i] - mn);
      float rs = p0 + p1;
#pragma unroll
      for (int d = 1; d < 16; d <<= 1) rs += __shfl_xor(rs, d, 32);
      lrow[i] = lrow[i] * alpha[i] + rs;
      mrow[i] = mn;
      int row = ((lane >> 4) << 3) + i;
      ps[row * 32 + (lane & 15)]      = (__bf16)p0;   // cols 0..15
      ps[row * 32 + 16 + (lane & 15)] = (__bf16)p1;   // cols 16..31
    }
#pragma unroll
    for (int j = 0; j < 4; ++j)
#pragma unroll
      for (int i = 0; i < 8; ++i) o[j][i] *= alpha[i];

    // P (16x32) * V_chunk (32x64); DS ops from one wave are in-order
    v16bf ap = load_A_bf16(ps, 32, lane);
    o[0] = wmma_bf16(ap, bv0, o[0]);
    o[1] = wmma_bf16(ap, bv1, o[1]);
    o[2] = wmma_bf16(ap, bv2, o[2]);
    o[3] = wmma_bf16(ap, bv3, o[3]);
  }

  int row = ((lane >> 4) << 3);
  int col = lane & 15;
  float rinv[8];
#pragma unroll
  for (int i = 0; i < 8; ++i) rinv[i] = 1.f / lrow[i];
#pragma unroll
  for (int j = 0; j < 4; ++j)
#pragma unroll
    for (int i = 0; i < 8; ++i)
      Ctx[(size_t)(tBase + row + i) * EDIM + h * HD + j * 16 + col] =
          (__bf16)(o[j][i] * rinv[i]);
  if (col == 0) {
#pragma unroll
    for (int i = 0; i < 8; ++i) {
      stat_m[(size_t)h * TDIM + tBase + row + i] = mrow[i];
      stat_l[(size_t)h * TDIM + tBase + row + i] = lrow[i];
    }
  }
}

// ---------------------------------------------------------------------------
// Kernel 3: averaged attention map. Recompute scores per (t,s) tile for all
// heads via WMMA, apply exp(s-m)/l, sum over heads, single coalesced write.
// Per-wave tile 32(t) x 64(s): each K-fragment pair feeds 4 WMMAs.
// Grid (T/256, S/64).
// ---------------------------------------------------------------------------
__global__ void __launch_bounds__(256)
attn_avg_kernel(const __bf16* __restrict__ Qb, const __bf16* __restrict__ Kb,
                const float* __restrict__ stat_m, const float* __restrict__ stat_l,
                float* __restrict__ Avg) {
  int lane = threadIdx.x & 31;
  int wave = threadIdx.x >> 5;
  int tBase = blockIdx.x * 256 + wave * 32;
  int sBase = blockIdx.y * 64;
  int col = lane & 15;

  v8f acc[2][4];
#pragma unroll
  for (int r = 0; r < 2; ++r)
#pragma unroll
    for (int j = 0; j < 4; ++j) acc[r][j] = v8f{0};

  for (int h = 0; h < HEADS; ++h) {
    const __bf16* Qh = Qb + (size_t)tBase * EDIM + h * HD;
    const __bf16* Kh = Kb + h * HD;
    v16bf aq00 = load_A_bf16(Qh, EDIM, lane);
    v16bf aq01 = load_A_bf16(Qh + 32, EDIM, lane);
    v16bf aq10 = load_A_bf16(Qh + 16 * EDIM, EDIM, lane);
    v16bf aq11 = load_A_bf16(Qh + 16 * EDIM + 32, EDIM, lane);
    float mi[2][8], li[2][8];
#pragma unroll
    for (int r = 0; r < 2; ++r) {
      int rbase = tBase + r * 16 + ((lane >> 4) << 3);
#pragma unroll
      for (int i = 0; i < 8; ++i) {
        mi[r][i] = stat_m[(size_t)h * TDIM + rbase + i];
        li[r][i] = 1.f / stat_l[(size_t)h * TDIM + rbase + i];
      }
    }
    v16bf b0 = load_Bt_bf16(Kh, EDIM, sBase, 0,  lane);
    v16bf b1 = load_Bt_bf16(Kh, EDIM, sBase, 32, lane);
#pragma unroll
    for (int j = 0; j < 4; ++j) {
      v16bf b0n = b0, b1n = b1;
      if (j < 3) {  // resolved statically (loop fully unrolled)
        b0n = load_Bt_bf16(Kh, EDIM, sBase + (j + 1) * 16, 0,  lane);
        b1n = load_Bt_bf16(Kh, EDIM, sBase + (j + 1) * 16, 32, lane);
      }
      v8f s0 = v8f{0}, s1 = v8f{0};
      s0 = wmma_bf16(aq00, b0, s0);
      s0 = wmma_bf16(aq01, b1, s0);
      s1 = wmma_bf16(aq10, b0, s1);
      s1 = wmma_bf16(aq11, b1, s1);
#pragma unroll
      for (int i = 0; i < 8; ++i) {
        acc[0][j][i] += __expf(s0[i] - mi[0][i]) * li[0][i];
        acc[1][j][i] += __expf(s1[i] - mi[1][i]) * li[1][i];
      }
      b0 = b0n; b1 = b1n;
    }
  }
  const float invH = 1.f / (float)HEADS;
#pragma unroll
  for (int r = 0; r < 2; ++r) {
    int rbase = tBase + r * 16 + ((lane >> 4) << 3);
#pragma unroll
    for (int j = 0; j < 4; ++j)
#pragma unroll
      for (int i = 0; i < 8; ++i)
        Avg[(size_t)(rbase + i) * SDIM + sBase + j * 16 + col] =
            acc[r][j][i] * invH;
  }
}

// ---------------------------------------------------------------------------
// Kernel 4: out-projection. out = ctx * Wout^T + bias (fp32 result).
// Per-wave 32(M) x 64(N) tile via the shared GEMM core.
// ---------------------------------------------------------------------------
__global__ void __launch_bounds__(256)
outproj_kernel(const __bf16* __restrict__ Ctx, const __bf16* __restrict__ W,
               const float* __restrict__ bias, float* __restrict__ Out) {
  int lane = threadIdx.x & 31;
  int wave = threadIdx.x >> 5;
  int mBase = blockIdx.x * 256 + wave * 32;
  int nBase = blockIdx.y * 64;
  const __bf16* Xp = Ctx + (size_t)mBase * EDIM;

  v8f c[2][4];
#pragma unroll
  for (int r = 0; r < 2; ++r)
#pragma unroll
    for (int j = 0; j < 4; ++j) c[r][j] = v8f{0};

  gemm_32x64(Xp, W, nBase, lane, c);

  int col = lane & 15;
#pragma unroll
  for (int r = 0; r < 2; ++r) {
    int row = mBase + r * 16 + ((lane >> 4) << 3);
#pragma unroll
    for (int j = 0; j < 4; ++j) {
      float bv = bias[nBase + j * 16 + col];
#pragma unroll
      for (int i = 0; i < 8; ++i)
        Out[(size_t)(row + i) * EDIM + nBase + j * 16 + col] = c[r][j][i] + bv;
    }
  }
}

// ---------------------------------------------------------------------------
extern "C" void kernel_launch(void* const* d_in, const int* in_sizes, int n_in,
                              void* d_out, int out_size, void* d_ws, size_t ws_size,
                              hipStream_t stream) {
  const float* q     = (const float*)d_in[0];
  const float* k     = (const float*)d_in[1];
  const float* v     = (const float*)d_in[2];
  const float* Wqkv  = (const float*)d_in[3];
  const float* bqkv  = (const float*)d_in[4];
  const float* Wout  = (const float*)d_in[5];
  const float* bout  = (const float*)d_in[6];

  float* out = (float*)d_out;                              // (T, 1, E)
  float* avg = out + (size_t)TDIM * EDIM;                  // (1, T, S)

  // Workspace layout (~64.5 MB total; fully L2-resident on MI455X)
  const size_t teBytes = (size_t)TDIM * EDIM * sizeof(__bf16);  // 8 MB
  char* w = (char*)d_ws;
  __bf16* Qb   = (__bf16*)(w);                 // (T x E) bf16, pre-scaled
  __bf16* Kb   = (__bf16*)(w + 1 * teBytes);   // (T x E) bf16
  __bf16* Vt   = (__bf16*)(w + 2 * teBytes);   // (E x T) bf16, transposed
  __bf16* Ctx  = (__bf16*)(w + 3 * teBytes);   // (T x E) bf16
  __bf16* Xqb  = (__bf16*)(w + 4 * teBytes);   // bf16 copies of inputs
  __bf16* Xkb  = (__bf16*)(w + 5 * teBytes);
  __bf16* Xvb  = (__bf16*)(w + 6 * teBytes);
  __bf16* Wb   = (__bf16*)(w + 7 * teBytes);               // (3E x E) bf16
  __bf16* Wob  = (__bf16*)((char*)Wb + (size_t)3 * EDIM * EDIM * 2);  // (E x E)
  float* stat_m = (float*)((char*)Wob + (size_t)EDIM * EDIM * 2);     // H*T
  float* stat_l = stat_m + (size_t)HEADS * TDIM;                      // H*T

  dim3 blk(256);
  const int nTE = TDIM * EDIM;
  cvt_bf16_kernel<<<nTE / 2048, blk, 0, stream>>>(q, Xqb, nTE);
  cvt_bf16_kernel<<<nTE / 2048, blk, 0, stream>>>(k, Xkb, nTE);
  cvt_bf16_kernel<<<nTE / 2048, blk, 0, stream>>>(v, Xvb, nTE);
  cvt_bf16_kernel<<<(3 * EDIM * EDIM) / 2048, blk, 0, stream>>>(Wqkv, Wb,
                                                                3 * EDIM * EDIM);
  cvt_bf16_kernel<<<(EDIM * EDIM) / 2048, blk, 0, stream>>>(Wout, Wob,
                                                            EDIM * EDIM);

  inproj_kernel<<<dim3(TDIM / 256, 3 * EDIM / 64), blk, 0, stream>>>(
      Xqb, Xkb, Xvb, Wb, bqkv, Qb, Kb, Vt);
  flash_attn_kernel<<<dim3(TDIM / 128, HEADS), blk, 0, stream>>>(
      Qb, Kb, Vt, Ctx, stat_m, stat_l);
  attn_avg_kernel<<<dim3(TDIM / 256, SDIM / 64), blk, 0, stream>>>(
      Qb, Kb, stat_m, stat_l, avg);
  outproj_kernel<<<dim3(TDIM / 256, EDIM / 64), blk, 0, stream>>>(
      Ctx, Wob, bout, out);
}
